// ROSA_QKV_LAYER_23510650978851
// MI455X (gfx1250) — matrix operand order repaired and checked by reading, not compiled, not run on hardware
//
#include <hip/hip_runtime.h>
#include <hip/hip_bf16.h>

#define TAU_INV 10.0f
#define TT      1024
#define DD      512
#define NH      8
#define HD      64
#define NB      2
#define NMAT    16   // NB*NH

typedef __attribute__((ext_vector_type(16))) __bf16 bf16x16;
typedef __attribute__((ext_vector_type(8)))  float  floatx8;

__device__ __forceinline__ floatx8 wmma_bf16f32(bf16x16 a, bf16x16 b, floatx8 c) {
#if defined(__gfx1250__)
    return __builtin_amdgcn_wmma_f32_16x16x32_bf16(false, a, false, b, (short)0, c, false, false);
#else
    (void)a; (void)b; return c;
#endif
}

// A-matrix tile (16x32 bf16): lanes 0-15 hold K=0..15 of row M=lane;
// lanes 16-31 hold K=16..31 (matches load_B contiguous order).
__device__ __forceinline__ bf16x16 load_A(const __bf16* A, int lda, int kb, int lane) {
    int m  = lane & 15;
    int k0 = kb + ((lane & 16) ? 16 : 0);
    union { bf16x16 v; uint4 u[2]; } r;
    r.u[0] = *(const uint4*)(A + (size_t)m * lda + k0);
    r.u[1] = *(const uint4*)(A + (size_t)m * lda + k0 + 8);
    return r.v;
}

// B-matrix tile (32x16 bf16) from BT stored row-major [N][K]: lane n holds 16
// consecutive K values of column n; lanes 0-15: K=kb..kb+15, lanes 16-31: +16.
__device__ __forceinline__ bf16x16 load_B(const __bf16* BT, int ldb, int kb, int lane) {
    int n  = lane & 15;
    int k0 = kb + ((lane & 16) ? 16 : 0);
    const uint4* p = (const uint4*)(BT + (size_t)n * ldb + k0);
    union { bf16x16 v; uint4 u[2]; } r;
    r.u[0] = p[0];
    r.u[1] = p[1];
    return r.v;
}

template<int NSETS, int KTOT>
__device__ __forceinline__ void gemm_tile(const __bf16* Abase, int lda,
                                          const __bf16* Bbase, int ldb,
                                          floatx8 (&acc)[NSETS], int lane) {
    for (int kk = 0; kk < KTOT; kk += 32) {
        bf16x16 a = load_A(Abase, lda, kk, lane);
#pragma unroll
        for (int s = 0; s < NSETS; ++s) {
            bf16x16 b = load_B(Bbase + (size_t)s * 16 * ldb, ldb, kk, lane);
            acc[s] = wmma_bf16f32(a, b, acc[s]);
        }
    }
}

// ---------------------------------------------------------------- repack ----
__global__ void cvt_x_kernel(const float* __restrict__ x, __bf16* __restrict__ xbf, int n) {
    int i = blockIdx.x * blockDim.x + threadIdx.x;
    if (i < n) xbf[i] = (__bf16)x[i];
}

// WT[n][k] = W[k][n]  (512x512)
__global__ void transpose_w_kernel(const float* __restrict__ w, __bf16* __restrict__ wT) {
    int i = blockIdx.x * blockDim.x + threadIdx.x;   // i over 512*512
    int k = i / DD, n = i % DD;                       // coalesced read
    wT[(size_t)n * DD + k] = (__bf16)w[(size_t)k * DD + n];
}

// woT[h][e][d] = wo[(h*64+d)*64 + e]
__global__ void pack_wo_kernel(const float* __restrict__ wo, __bf16* __restrict__ woT) {
    int i = blockIdx.x * blockDim.x + threadIdx.x;   // over 8*64*64
    int h = i >> 12, rem = i & 4095;
    int e = rem >> 6, d = rem & 63;
    woT[i] = (__bf16)wo[((size_t)(h * HD + d)) * HD + e];
}

// ----------------------------------------------------- QKV + head softmax ---
// wave -> (mat in {q,k,v}, head h, 16-row tile). 16x64 f32 accum, softmax over
// the 64 head columns in-register, store q/k row-major [bh][t][d], v transposed.
__global__ void qkv_kernel(const __bf16* __restrict__ xbf, const __bf16* __restrict__ wT,
                           __bf16* __restrict__ q, __bf16* __restrict__ k,
                           __bf16* __restrict__ vT) {
    int gid  = blockIdx.x * blockDim.x + threadIdx.x;
    int w    = gid >> 5, lane = gid & 31;
    int mat  = w / (NH * 128);
    int rem  = w % (NH * 128);
    int h    = rem / 128;
    int rt   = rem % 128;
    int row0 = rt * 16;

    const __bf16* A  = xbf + (size_t)row0 * DD;
    const __bf16* Bb = wT + (size_t)mat * DD * DD + (size_t)h * HD * DD;
    floatx8 acc[4] = {};
    gemm_tile<4, DD>(A, DD, Bb, DD, acc, lane);

#pragma unroll
    for (int j = 0; j < 8; ++j) {
        float m = fmaxf(fmaxf(acc[0][j], acc[1][j]), fmaxf(acc[2][j], acc[3][j]));
        for (int off = 8; off >= 1; off >>= 1) m = fmaxf(m, __shfl_xor(m, off, 32));
        float ssum = 0.f;
#pragma unroll
        for (int s = 0; s < 4; ++s) {
            float e = __expf((acc[s][j] - m) * TAU_INV);
            acc[s][j] = e; ssum += e;
        }
        for (int off = 8; off >= 1; off >>= 1) ssum += __shfl_xor(ssum, off, 32);
        float inv = 1.0f / ssum;
#pragma unroll
        for (int s = 0; s < 4; ++s) acc[s][j] *= inv;
    }

    int n  = lane & 15;
    int b  = row0 >> 10;
    int t0 = row0 & (TT - 1);
    int bh = b * NH + h;
#pragma unroll
    for (int s = 0; s < 4; ++s)
#pragma unroll
        for (int j = 0; j < 8; ++j) {
            int mrow = (lane & 16) ? (j + 8) : j;
            int tt   = t0 + mrow;
            int d    = s * 16 + n;
            __bf16 val = (__bf16)acc[s][j];
            if (mat == 0)      q [((size_t)bh * TT + tt) * HD + d] = val;
            else if (mat == 1) k [((size_t)bh * TT + tt) * HD + d] = val;
            else               vT[((size_t)bh * HD + d) * TT + tt] = val;
        }
}

// --------------------------------------------------------- S = q @ k^T ------
__global__ void score_kernel(const __bf16* __restrict__ q, const __bf16* __restrict__ k,
                             float* __restrict__ S) {
    int gid = blockIdx.x * blockDim.x + threadIdx.x;
    int w = gid >> 5, lane = gid & 31;
    int bh  = w / (64 * 16);
    int rem = w % (64 * 16);
    int rt = rem / 16, ct = rem % 16;

    const __bf16* A  = q + ((size_t)bh * TT + rt * 16) * HD;
    const __bf16* Bb = k + ((size_t)bh * TT + ct * 64) * HD;   // BT rows = k rows
    floatx8 acc[4] = {};
    gemm_tile<4, HD>(A, HD, Bb, HD, acc, lane);

    int n = lane & 15;
    float* Sm = S + ((size_t)bh << 20);
#pragma unroll
    for (int s = 0; s < 4; ++s)
#pragma unroll
        for (int j = 0; j < 8; ++j) {
            int mrow = (lane & 16) ? (j + 8) : j;
            Sm[(size_t)(rt * 16 + mrow) * TT + ct * 64 + s * 16 + n] = acc[s][j];
        }
}

// -------------------------------------------- diagonal recurrence (in place) -
// y[r,c] = S[r,c]*(y[r-1,c-1]+1). Wave owns 32 consecutive diagonals; walking
// rows keeps each iteration's loads/stores contiguous (lane reads col r-d).
__global__ void scan_kernel(float* __restrict__ S) {
    int gid = blockIdx.x * blockDim.x + threadIdx.x;
    int w = gid >> 5, lane = gid & 31;
    int mat = w >> 5;
    int d0  = (w & 31) * 32;
    int d   = d0 + lane;
    float* M = S + ((size_t)mat << 20);
    float y = 0.0f;
    for (int r = d0; r < TT; ++r) {
        if (r >= d) {
            int c = r - d;
            float xv = M[(size_t)r * TT + c];
            y = xv * (y + 1.0f);
            M[(size_t)r * TT + c] = y;
        }
    }
}

// --------------------------------- causal softmax with +c/(r+1) bias, tau ----
__global__ void attnsm_kernel(const float* __restrict__ S, __bf16* __restrict__ P) {
    int gid = blockIdx.x * blockDim.x + threadIdx.x;
    int w = gid >> 5, lane = gid & 31;
    int mat = w >> 10;
    int r   = w & (TT - 1);
    const float* row = S + ((size_t)mat << 20) + (size_t)r * TT;
    float rinv = 1.0f / (float)(r + 1);

    float zv[TT / 32];
    float m = -3.0e38f;
#pragma unroll
    for (int i = 0; i < TT / 32; ++i) {
        int c = i * 32 + lane;
        float z = -3.0e38f;
        if (c <= r) z = row[c] + (float)c * rinv;
        zv[i] = z;
        m = fmaxf(m, z);
    }
    for (int off = 16; off >= 1; off >>= 1) m = fmaxf(m, __shfl_xor(m, off, 32));
    float ssum = 0.f;
#pragma unroll
    for (int i = 0; i < TT / 32; ++i) {
        int c = i * 32 + lane;
        float e = 0.f;
        if (c <= r) e = __expf((zv[i] - m) * TAU_INV);
        zv[i] = e; ssum += e;
    }
    for (int off = 16; off >= 1; off >>= 1) ssum += __shfl_xor(ssum, off, 32);
    float inv = 1.0f / ssum;
    __bf16* prow = P + ((size_t)mat << 20) + (size_t)r * TT;
#pragma unroll
    for (int i = 0; i < TT / 32; ++i) prow[i * 32 + lane] = (__bf16)(zv[i] * inv);
}

// ------------------------------------------------------------ O = P @ V -----
__global__ void pv_kernel(const __bf16* __restrict__ P, const __bf16* __restrict__ vT,
                          __bf16* __restrict__ o) {
    int gid = blockIdx.x * blockDim.x + threadIdx.x;
    int w = gid >> 5, lane = gid & 31;
    int bh = w / 64, rt = w % 64;

    const __bf16* A  = P + ((size_t)bh << 20) + (size_t)(rt * 16) * TT;
    const __bf16* Bb = vT + (size_t)bh * HD * TT;   // BT rows = V^T rows
    floatx8 acc[4] = {};
    gemm_tile<4, TT>(A, TT, Bb, TT, acc, lane);

    int n = lane & 15;
#pragma unroll
    for (int s = 0; s < 4; ++s)
#pragma unroll
        for (int j = 0; j < 8; ++j) {
            int mrow = (lane & 16) ? (j + 8) : j;
            o[((size_t)bh * TT + rt * 16 + mrow) * HD + s * 16 + n] = (__bf16)acc[s][j];
        }
}

// ------------------------------------------ out = o @ wo_h, interleave heads -
__global__ void proj_kernel(const __bf16* __restrict__ o, const __bf16* __restrict__ woT,
                            float* __restrict__ out) {
    int gid = blockIdx.x * blockDim.x + threadIdx.x;
    int w = gid >> 5, lane = gid & 31;
    int bh = w / 64, rt = w % 64;
    int b = bh >> 3, h = bh & 7;

    const __bf16* A  = o + ((size_t)bh * TT + rt * 16) * HD;
    const __bf16* Bb = woT + (size_t)h * HD * HD;
    floatx8 acc[4] = {};
    gemm_tile<4, HD>(A, HD, Bb, HD, acc, lane);

    int n = lane & 15;
#pragma unroll
    for (int s = 0; s < 4; ++s)
#pragma unroll
        for (int j = 0; j < 8; ++j) {
            int mrow = (lane & 16) ? (j + 8) : j;
            out[((size_t)(b * TT + rt * 16 + mrow)) * DD + h * HD + s * 16 + n] = acc[s][j];
        }
}

// ---------------------------------------------------------------- launch ----
extern "C" void kernel_launch(void* const* d_in, const int* in_sizes, int n_in,
                              void* d_out, int out_size, void* d_ws, size_t ws_size,
                              hipStream_t stream) {
    const float* x  = (const float*)d_in[0];
    const float* wq = (const float*)d_in[1];
    const float* wk = (const float*)d_in[2];
    const float* wv = (const float*)d_in[3];
    const float* wo = (const float*)d_in[4];
    float* out = (float*)d_out;

    char* ws = (char*)d_ws;
    size_t off = 0;
    auto carve = [&](size_t bytes) { char* p = ws + off; off = (off + bytes + 255) & ~(size_t)255; return p; };

    __bf16* xbf = (__bf16*)carve((size_t)NB * TT * DD * 2);
    __bf16* wT  = (__bf16*)carve((size_t)3 * DD * DD * 2);
    __bf16* woT = (__bf16*)carve((size_t)NH * HD * HD * 2);
    __bf16* q   = (__bf16*)carve((size_t)NMAT * TT * HD * 2);
    __bf16* k   = (__bf16*)carve((size_t)NMAT * TT * HD * 2);
    __bf16* vT  = (__bf16*)carve((size_t)NMAT * HD * TT * 2);
    float*  S   = (float*) carve((size_t)NMAT * TT * TT * 4);
    __bf16* P   = (__bf16*)carve((size_t)NMAT * TT * TT * 2);
    __bf16* o   = (__bf16*)carve((size_t)NMAT * TT * HD * 2);
    (void)ws_size; (void)in_sizes; (void)n_in; (void)out_size;

    const int BLK = 256;
    cvt_x_kernel      <<<(NB * TT * DD) / BLK, BLK, 0, stream>>>(x, xbf, NB * TT * DD);
    transpose_w_kernel<<<(DD * DD) / BLK, BLK, 0, stream>>>(wq, wT);
    transpose_w_kernel<<<(DD * DD) / BLK, BLK, 0, stream>>>(wk, wT + (size_t)DD * DD);
    transpose_w_kernel<<<(DD * DD) / BLK, BLK, 0, stream>>>(wv, wT + (size_t)2 * DD * DD);
    pack_wo_kernel    <<<(NH * HD * HD) / BLK, BLK, 0, stream>>>(wo, woT);

    qkv_kernel   <<<(3 * NH * 128 * 32) / BLK, BLK, 0, stream>>>(xbf, wT, q, k, vT);
    score_kernel <<<(NMAT * 64 * 16 * 32) / BLK, BLK, 0, stream>>>(q, k, S);
    scan_kernel  <<<(NMAT * 32 * 32) / BLK, BLK, 0, stream>>>(S);
    attnsm_kernel<<<(NMAT * TT * 32) / BLK, BLK, 0, stream>>>(S, P);
    pv_kernel    <<<(NMAT * 64 * 32) / BLK, BLK, 0, stream>>>(P, vT, o);
    proj_kernel  <<<(NMAT * 64 * 32) / BLK, BLK, 0, stream>>>(o, woT, out);
}
